// GraphAttentionEncoder_17970143166678
// MI455X (gfx1250) — matrix operand order repaired and checked by reading, not compiled, not run on hardware
//
#include <hip/hip_runtime.h>
#include <hip/hip_bf16.h>

// ---------------------------------------------------------------------------
// GAT encoder for MI455X (gfx1250, wave32).
// GEMMs run on v_wmma_f32_16x16x32_bf16 (fp32 data converted to bf16 RNE on
// the fly while staging into LDS). Attention softmax/scatter use f32 atomics
// (all traffic is L2-resident: weights 16.8MB, activations 82MB << 192MB L2).
// ---------------------------------------------------------------------------

typedef __attribute__((ext_vector_type(16))) __bf16 v16bf;
typedef __attribute__((ext_vector_type(8)))  float  v8f;

__device__ __forceinline__ unsigned pk_bf16(float a, float b) {
    unsigned ua = __float_as_uint(a), ub = __float_as_uint(b);
    ua = (ua + 0x7FFFu + ((ua >> 16) & 1u)) >> 16;   // round-to-nearest-even
    ub = (ub + 0x7FFFu + ((ub >> 16) & 1u)) >> 16;
    return (ub << 16) | (ua & 0xFFFFu);
}

// monotone float->uint key for atomic segment-max
__device__ __forceinline__ unsigned fkey(float v) {
    unsigned u = __float_as_uint(v);
    return (u & 0x80000000u) ? ~u : (u | 0x80000000u);
}
__device__ __forceinline__ float fkeyinv(unsigned k) {
    unsigned u = (k & 0x80000000u) ? (k & 0x7FFFFFFFu) : ~k;
    return __uint_as_float(u);
}

// ---------------------------------------------------------------------------
// WMMA GEMM: C[M,N] = epi( A[M,K] @ W[K,N] ), fp32 in/out, bf16 compute.
// Block = 256 threads = 8 waves (2x4), block tile 64x128, wave tile 32x32.
// Requires K % 32 == 0, N % 128 == 0 (holds: K in {768,2048,256}, N in
// {2048,256}). M edge (10000 % 64) handled by guards/zero-fill.
// epi: 0 = none, 1 = bias + relu.
// ---------------------------------------------------------------------------
__global__ __launch_bounds__(256)
void wmma_gemm(const float* __restrict__ A, const float* __restrict__ W,
               const float* __restrict__ bias, float* __restrict__ C,
               int M, int K, int N, int epi)
{
    // packed bf16 pairs; row stride 24 uints (96B) keeps 16B alignment for b128
    __shared__ __align__(16) unsigned lsA[64][24];   // [row][kpair]
    __shared__ __align__(16) unsigned lsB[128][24];  // [col][kpair]

    const int tid  = threadIdx.x;
    const int lane = tid & 31;
    const int wv   = tid >> 5;        // wave 0..7
    const int wm   = wv >> 2;         // 0..1  (M direction)
    const int wn   = wv & 3;          // 0..3  (N direction)
    const int mBase = blockIdx.y * 64;
    const int nBase = blockIdx.x * 128;
    const int ln16  = lane & 15;
    const int hiH   = lane >> 4;      // lane half selects K sub-block

    v8f acc[2][2] = {};

    for (int k0 = 0; k0 < K; k0 += 32) {
        // ---- stage A tile: 64 rows x 16 k-pairs (coalesced, zero-fill OOB)
        #pragma unroll
        for (int i = 0; i < 4; ++i) {
            int p = tid + i * 256;
            int kp = p & 15, row = p >> 4;
            int g = mBase + row;
            float a0 = 0.f, a1 = 0.f;
            if (g < M) {
                const float* ap = A + (size_t)g * K + k0 + 2 * kp;
                a0 = ap[0]; a1 = ap[1];
            }
            lsA[row][kp] = pk_bf16(a0, a1);
        }
        // ---- stage B tile: 128 cols x 16 k-pairs (consecutive tid -> consecutive n)
        #pragma unroll
        for (int i = 0; i < 8; ++i) {
            int p = tid + i * 256;
            int n = p & 127, kp = p >> 7;
            const float* wp = W + (size_t)(k0 + 2 * kp) * N + nBase + n;
            lsB[n][kp] = pk_bf16(wp[0], wp[N]);
        }
        __syncthreads();

        // ---- fragments per ISA 16-bit A(16x32)/B(32x16) wave32 layouts
        union Frag { v16bf v; uint4 q[2]; } afr[2], bfr[2];
        #pragma unroll
        for (int mi = 0; mi < 2; ++mi) {
            int row = wm * 32 + mi * 16 + ln16;
            int ka = hiH ? 4 : 0;                       // K 0-7 / 8-15
            afr[mi].q[0] = *(const uint4*)&lsA[row][ka];
            afr[mi].q[1] = *(const uint4*)&lsA[row][8 + ka]; // K 16-23 / 24-31
        }
        #pragma unroll
        for (int ni = 0; ni < 2; ++ni) {
            int col = wn * 32 + ni * 16 + ln16;
            int kb = hiH ? 4 : 0;                       // K 0-7 / 8-15, then 16-23 / 24-31
            bfr[ni].q[0] = *(const uint4*)&lsB[col][kb];
            bfr[ni].q[1] = *(const uint4*)&lsB[col][kb + 8];
        }
        #pragma unroll
        for (int mi = 0; mi < 2; ++mi)
            #pragma unroll
            for (int ni = 0; ni < 2; ++ni)
                acc[mi][ni] = __builtin_amdgcn_wmma_f32_16x16x32_bf16(
                    false, afr[mi].v, false, bfr[ni].v,
                    (short)0, acc[mi][ni], false, false);

        __syncthreads();
    }

    // ---- store per f32 C/D lane map: VGPR i -> M = i + 8*hiH, N = lane%16
    #pragma unroll
    for (int mi = 0; mi < 2; ++mi)
        #pragma unroll
        for (int ni = 0; ni < 2; ++ni)
            #pragma unroll
            for (int i = 0; i < 8; ++i) {
                int m = mBase + wm * 32 + mi * 16 + i + hiH * 8;
                int n = nBase + wn * 32 + ni * 16 + ln16;
                if (m < M) {
                    float v = acc[mi][ni][i];
                    if (epi == 1) { v += bias[n]; v = v > 0.f ? v : 0.f; }
                    C[(size_t)m * N + n] = v;
                }
            }
}

// ---------------------------------------------------------------------------
// alpha_s/alpha_d: per-node per-head dot(h[n,hd,:], a_src/a_dst[hd,:]).
// One block per node; one wave (32 lanes) per head (8 heads = 256 threads).
// ---------------------------------------------------------------------------
__global__ __launch_bounds__(256)
void k_alpha(const float* __restrict__ h, const float* __restrict__ a_src,
             const float* __restrict__ a_dst, float* __restrict__ as,
             float* __restrict__ ad)
{
    int n    = blockIdx.x;
    int hd   = threadIdx.x >> 5;
    int lane = threadIdx.x & 31;
    const float* hp = h + (size_t)n * 2048 + hd * 256;
    const float* sp = a_src + hd * 256;
    const float* dp = a_dst + hd * 256;
    float vs = 0.f, vd = 0.f;
    #pragma unroll
    for (int c = lane; c < 256; c += 32) {
        float x = hp[c];
        vs += x * sp[c]; vd += x * dp[c];
    }
    #pragma unroll
    for (int off = 16; off; off >>= 1) {
        vs += __shfl_down(vs, off, 32);
        vd += __shfl_down(vd, off, 32);
    }
    if (lane == 0) { as[n * 8 + hd] = vs; ad[n * 8 + hd] = vd; }
}

__device__ __forceinline__ void edge_endpoints(const long long* __restrict__ ei,
                                               int E0, int e, int& s, int& d)
{
    if (e < E0) { s = (int)ei[e]; d = (int)ei[E0 + e]; }
    else        { s = d = e - E0; }   // appended self-loops
}

// e = leaky_relu(as[src] + ad[dst]); atomic segment-max into mkey[dst]
__global__ void k_edge_score(const long long* __restrict__ ei, int E0, int N,
                             const float* __restrict__ as, const float* __restrict__ ad,
                             float* __restrict__ ebuf, unsigned* __restrict__ mkey)
{
    int idx = blockIdx.x * blockDim.x + threadIdx.x;
    int E = E0 + N;
    if (idx >= E * 8) return;
    int e = idx >> 3, hd = idx & 7, s, d;
    edge_endpoints(ei, E0, e, s, d);
    float v = as[s * 8 + hd] + ad[d * 8 + hd];
    v = v > 0.f ? v : 0.2f * v;
    ebuf[idx] = v;
    atomicMax(&mkey[d * 8 + hd], fkey(v));
}

// p = exp(e - max[dst]); denom[dst] += p (every dst has a self-loop => finite max)
__global__ void k_edge_exp(const long long* __restrict__ ei, int E0, int N,
                           float* __restrict__ ebuf, const unsigned* __restrict__ mkey,
                           float* __restrict__ denom)
{
    int idx = blockIdx.x * blockDim.x + threadIdx.x;
    int E = E0 + N;
    if (idx >= E * 8) return;
    int e = idx >> 3, hd = idx & 7, s, d;
    edge_endpoints(ei, E0, e, s, d);
    float m = fkeyinv(mkey[d * 8 + hd]);
    float p = __expf(ebuf[idx] - m);
    ebuf[idx] = p;
    atomicAdd(&denom[d * 8 + hd], p);
}

// out[dst, hd*256+c] += alpha * h[src, hd*256+c]; one block per edge
__global__ __launch_bounds__(256)
void k_edge_aggr(const long long* __restrict__ ei, int E0, int N,
                 const float* __restrict__ ebuf, const float* __restrict__ denom,
                 const float* __restrict__ h, float* __restrict__ out)
{
    int e = blockIdx.x, s, d;
    edge_endpoints(ei, E0, e, s, d);
    __shared__ float alpha[8];
    int t = threadIdx.x;
    if (t < 8) alpha[t] = ebuf[e * 8 + t] / (denom[d * 8 + t] + 1e-16f);
    __syncthreads();
    const float* hs = h + (size_t)s * 2048;
    float* od = out + (size_t)d * 2048;
    #pragma unroll
    for (int hd = 0; hd < 8; ++hd) {
        int j = hd * 256 + t;
        atomicAdd(&od[j], alpha[hd] * hs[j]);
    }
}

// in-place: x = elu(x + b1[j])   (layer-1 epilogue)
__global__ void k_elu_bias(float* __restrict__ x, const float* __restrict__ b,
                           int total, int HC)
{
    int idx = blockIdx.x * blockDim.x + threadIdx.x;
    if (idx >= total) return;
    float v = x[idx] + b[idx % HC];
    x[idx] = v > 0.f ? v : (__expf(v) - 1.f);
}

// mean over 8 heads + b2 : [N,2048] -> [N,256]
__global__ __launch_bounds__(256)
void k_head_mean(const float* __restrict__ in, const float* __restrict__ b,
                 float* __restrict__ out)
{
    int n = blockIdx.x, c = threadIdx.x;
    const float* p = in + (size_t)n * 2048;
    float s = 0.f;
    #pragma unroll
    for (int hd = 0; hd < 8; ++hd) s += p[hd * 256 + c];
    out[n * 256 + c] = 0.125f * s + b[c];
}

// ---------------------------------------------------------------------------
extern "C" void kernel_launch(void* const* d_in, const int* in_sizes, int n_in,
                              void* d_out, int out_size, void* d_ws, size_t ws_size,
                              hipStream_t stream)
{
    const float*     x   = (const float*)d_in[0];
    const long long* ei  = (const long long*)d_in[1];   // int64 per reference
    const float*     W1  = (const float*)d_in[2];
    const float*     aS1 = (const float*)d_in[3];
    const float*     aD1 = (const float*)d_in[4];
    const float*     b1  = (const float*)d_in[5];
    const float*     W2  = (const float*)d_in[6];
    const float*     aS2 = (const float*)d_in[7];
    const float*     aD2 = (const float*)d_in[8];
    const float*     b2  = (const float*)d_in[9];
    const float*     Wp  = (const float*)d_in[10];
    const float*     bp  = (const float*)d_in[11];

    const int N = 10000, E0 = 80000, HC = 2048, C = 256, IN = 768;
    const int E = E0 + N;

    // workspace layout (~179 MB)
    float*    bufA  = (float*)d_ws;                         // h1 / h2      [N*HC]
    float*    bufB  = bufA + (size_t)N * HC;                // out1 / out2  [N*HC]
    float*    asb   = bufB + (size_t)N * HC;                // [N*8]
    float*    adb   = asb + (size_t)N * 8;                  // [N*8]
    unsigned* mkey  = (unsigned*)(adb + (size_t)N * 8);     // [N*8]
    float*    denom = (float*)(mkey + (size_t)N * 8);       // [N*8]
    float*    ebuf  = denom + (size_t)N * 8;                // [E*8]
    float*    meanb = ebuf + (size_t)E * 8;                 // [N*C]

    dim3 blk(256);
    const int mt = (N + 63) / 64;
    const int edgeThreads = E * 8;
    const int edgeBlocks  = (edgeThreads + 255) / 256;

    // ---------------- layer 1 ----------------
    wmma_gemm<<<dim3(HC / 128, mt), blk, 0, stream>>>(x, W1, nullptr, bufA, N, IN, HC, 0);
    k_alpha<<<N, blk, 0, stream>>>(bufA, aS1, aD1, asb, adb);

    hipMemsetAsync(mkey,  0, (size_t)N * 8 * 4, stream);
    hipMemsetAsync(denom, 0, (size_t)N * 8 * 4, stream);
    hipMemsetAsync(bufB,  0, (size_t)N * HC * 4, stream);

    k_edge_score<<<edgeBlocks, blk, 0, stream>>>(ei, E0, N, asb, adb, ebuf, mkey);
    k_edge_exp  <<<edgeBlocks, blk, 0, stream>>>(ei, E0, N, ebuf, mkey, denom);
    k_edge_aggr <<<E, blk, 0, stream>>>(ei, E0, N, ebuf, denom, bufA, bufB);
    k_elu_bias  <<<((size_t)N * HC + 255) / 256, blk, 0, stream>>>(bufB, b1, N * HC, HC);

    // ---------------- layer 2 ----------------
    wmma_gemm<<<dim3(HC / 128, mt), blk, 0, stream>>>(bufB, W2, nullptr, bufA, N, HC, HC, 0);
    k_alpha<<<N, blk, 0, stream>>>(bufA, aS2, aD2, asb, adb);

    hipMemsetAsync(mkey,  0, (size_t)N * 8 * 4, stream);
    hipMemsetAsync(denom, 0, (size_t)N * 8 * 4, stream);
    hipMemsetAsync(bufB,  0, (size_t)N * HC * 4, stream);

    k_edge_score<<<edgeBlocks, blk, 0, stream>>>(ei, E0, N, asb, adb, ebuf, mkey);
    k_edge_exp  <<<edgeBlocks, blk, 0, stream>>>(ei, E0, N, ebuf, mkey, denom);
    k_edge_aggr <<<E, blk, 0, stream>>>(ei, E0, N, ebuf, denom, bufA, bufB);
    k_head_mean <<<N, blk, 0, stream>>>(bufB, b2, meanb);

    // ---------------- projection + relu -> d_out ----------------
    wmma_gemm<<<dim3(C / 128, mt), blk, 0, stream>>>(meanb, Wp, bp, (float*)d_out,
                                                     N, C, C, 1);
}
